// LineAttention_71253507440750
// MI455X (gfx1250) — matrix operand order repaired and checked by reading, not compiled, not run on hardware
//
#include <hip/hip_runtime.h>

#define LINE   24
#define HEADS  8
#define DHEAD  64
#define DIM    256
#define INNER  512

typedef _Float16 v16h __attribute__((ext_vector_type(16)));
typedef _Float16 v8h  __attribute__((ext_vector_type(8)));
typedef float    v8f  __attribute__((ext_vector_type(8)));

// ---------------- LDS layout (bytes) ----------------
// xh   [32][264] f16 = 16896  (aliased later by attn [8][32][32] f16 = 16384)
// q    [32][520] f16 = 33280  (aliased later by out  [32][520] f16)
// k    [32][520] f16 = 33280
// vT   [512][40] f16 = 40960
// sim  [8][32][32] f32 = 32768
#define XH_PITCH 264
#define QO_PITCH 520
#define VT_PITCH 40
#define OFF_XH   0
#define OFF_Q    16896
#define OFF_K    50176
#define OFF_VT   83456
#define OFF_SIM  124416
#define SMEM_BYTES 157184

__device__ __forceinline__ v8f vzero() {
  v8f z = {0.f,0.f,0.f,0.f,0.f,0.f,0.f,0.f};
  return z;
}

// A-fragment (16x32 f16): lane l holds row l%16, K in [16*(l/16), +16) contiguous.
// Caller passes p = rowbase + 16*(l>>4); we fetch p[0..7] and p[8..15].
__device__ __forceinline__ v16h ldA(const _Float16* p) {
  v16h f;
  reinterpret_cast<uint4*>(&f)[0] = *reinterpret_cast<const uint4*>(p);
  reinterpret_cast<uint4*>(&f)[1] = *reinterpret_cast<const uint4*>(p + 8);
  return f;
}

// B-fragment (32x16 f16): lane l holds column l%16, K in [16*(l/16), +16) contiguous.
// Caller passes p = colrow_base + 16*(l>>4); we fetch 16 contiguous halves.
__device__ __forceinline__ v16h ldB(const _Float16* p) {
  v16h f;
  reinterpret_cast<uint4*>(&f)[0] = reinterpret_cast<const uint4*>(p)[0];
  reinterpret_cast<uint4*>(&f)[1] = reinterpret_cast<const uint4*>(p)[1];
  return f;
}

__device__ __forceinline__ v8f wmma(v16h a, v16h b, v8f c) {
  return __builtin_amdgcn_wmma_f32_16x16x32_f16(
      /*neg_a=*/false, a, /*neg_b=*/false, b,
      /*c_mod=*/(short)0, c, /*reuse_a=*/false, /*reuse_b=*/false);
}

// ---------------- weight prep: fp32 -> transposed f16 ----------------
// W1h [1536][256]: n<512 -> Wq[:,n]^T ; n>=512 -> Wkv[:,n-512]^T (covers both k and v halves)
// Woh [256][512]:  Wo^T
__global__ void prep_kernel(const float* __restrict__ Wq, const float* __restrict__ Wkv,
                            const float* __restrict__ Wo, _Float16* __restrict__ W1h,
                            _Float16* __restrict__ Woh) {
  int idx = blockIdx.x * 256 + threadIdx.x;
  if (idx < 1536 * 256) {
    int n = idx >> 8, k = idx & 255;
    float v = (n < 512) ? Wq[k * 512 + n] : Wkv[k * 1024 + (n - 512)];
    W1h[idx] = (_Float16)v;
  } else {
    int j = idx - 1536 * 256;          // 0 .. 131071
    int n = j >> 9, k = j & 511;
    Woh[j] = (_Float16)Wo[k * 256 + n];
  }
}

// ---------------- main: one workgroup per line (256 threads = 8 waves) ----------------
__global__ void __launch_bounds__(256, 1) line_attn_kernel(
    const float* __restrict__ x, const float* __restrict__ bo,
    const float* __restrict__ pos, const _Float16* __restrict__ W1h,
    const _Float16* __restrict__ Woh, float* __restrict__ y) {
  extern __shared__ uint4 smem4[];
  char* smem = (char*)smem4;
  _Float16* xh   = (_Float16*)(smem + OFF_XH);
  _Float16* attn = (_Float16*)(smem + OFF_XH);   // alias: xh dead after GEMM1
  _Float16* qb   = (_Float16*)(smem + OFF_Q);
  _Float16* outb = (_Float16*)(smem + OFF_Q);    // alias: q dead after sim
  _Float16* kb   = (_Float16*)(smem + OFF_K);
  _Float16* vT   = (_Float16*)(smem + OFF_VT);
  float*    sim  = (float*)(smem + OFF_SIM);

  const int tid = threadIdx.x;
  const int wv  = tid >> 5;        // wave id 0..7
  const int l   = tid & 31;        // lane
  const int lm  = l & 15;
  const int lh  = l >> 4;

  // ---- Phase 1: load x line -> f16 LDS, zero pad rows 24..31 ----
  const float* xline = x + (size_t)blockIdx.x * (LINE * DIM);
  for (int e = tid; e < LINE * DIM; e += 256) {
    int r = e >> 8, c = e & 255;
    xh[r * XH_PITCH + c] = (_Float16)xline[e];
  }
  for (int e = tid; e < 8 * DIM; e += 256) {
    int r = LINE + (e >> 8), c = e & 255;
    xh[r * XH_PITCH + c] = (_Float16)0.f;
  }
  __syncthreads();

  // ---- Phase 2: GEMM1  [32x256] x [256x1536] -> q | k | vT ----
  // A (the x tile) is register-resident for the whole loop: 2 Mtiles x 8 ksteps.
  {
    v16h areg[2][8];
#pragma unroll
    for (int m = 0; m < 2; ++m)
#pragma unroll
      for (int ks = 0; ks < 8; ++ks)
        areg[m][ks] = ldA(xh + (16 * m + lm) * XH_PITCH + 32 * ks + 16 * lh);

    for (int it = 0; it < 12; ++it) {
      const int nt = it * 8 + wv;                  // N-tile 0..95
      v8f acc0 = vzero(), acc1 = vzero();
      const _Float16* wrow = W1h + (nt * 16 + lm) * 256 + 16 * lh;
#pragma unroll
      for (int ks = 0; ks < 8; ++ks) {
        v16h b = ldB(wrow + 32 * ks);
        acc0 = wmma(areg[0][ks], b, acc0);
        acc1 = wmma(areg[1][ks], b, acc1);
      }
      if (nt < 64) {                                // q (scaled) or k, row-major
        _Float16* dst = (nt < 32) ? qb : kb;
        const int   c0  = (nt < 32) ? nt * 16 : (nt - 32) * 16;
        const float scl = (nt < 32) ? 0.125f : 1.0f;   // DHEAD^-0.5
#pragma unroll
        for (int r = 0; r < 8; ++r) {
          dst[(8 * lh + r) * QO_PITCH + c0 + lm]        = (_Float16)(acc0[r] * scl);
          dst[(16 + 8 * lh + r) * QO_PITCH + c0 + lm]   = (_Float16)(acc1[r] * scl);
        }
      } else {                                      // v stored transposed: vT[d][token]
        const int d0 = (nt - 64) * 16;
        v8h p0, p1;
#pragma unroll
        for (int r = 0; r < 8; ++r) { p0[r] = (_Float16)acc0[r]; p1[r] = (_Float16)acc1[r]; }
        *reinterpret_cast<uint4*>(vT + (d0 + lm) * VT_PITCH + 8 * lh)      = *reinterpret_cast<uint4*>(&p0);
        *reinterpret_cast<uint4*>(vT + (d0 + lm) * VT_PITCH + 16 + 8 * lh) = *reinterpret_cast<uint4*>(&p1);
      }
    }
  }
  __syncthreads();

  // ---- Phase 3: sim = q . k^T per head (wave == head) ----
  {
    const int h = wv;
    float* simh = sim + h * 1024;
    const _Float16* qh = qb + 64 * h;
    const _Float16* kh = kb + 64 * h;
#pragma unroll
    for (int mt = 0; mt < 2; ++mt)
#pragma unroll
      for (int ntj = 0; ntj < 2; ++ntj) {
        v8f acc = vzero();
#pragma unroll
        for (int ks = 0; ks < 2; ++ks) {
          v16h a = ldA(qh + (16 * mt + lm) * QO_PITCH + 32 * ks + 16 * lh);
          v16h b = ldB(kh + (16 * ntj + lm) * QO_PITCH + 32 * ks + 16 * lh);
          acc = wmma(a, b, acc);
        }
#pragma unroll
        for (int r = 0; r < 8; ++r)
          simh[(16 * mt + 8 * lh + r) * 32 + 16 * ntj + lm] = acc[r];
      }
  }
  __syncthreads();

  // ---- Phase 4: softmax(sim + pos_emb) -> attn (f16), pad zeroed ----
  if (tid < HEADS * LINE) {
    const int h = tid / LINE, i = tid - h * LINE;
    const float* srow = sim + h * 1024 + i * 32;
    const float* prow = pos + (h * LINE + i) * LINE;
    float v[LINE];
    float m = -1e30f;
#pragma unroll
    for (int j = 0; j < LINE; ++j) { v[j] = srow[j] + prow[j]; m = fmaxf(m, v[j]); }
    float s = 0.f;
#pragma unroll
    for (int j = 0; j < LINE; ++j) { float e = __expf(v[j] - m); v[j] = e; s += e; }
    const float inv = 1.0f / s;
    _Float16* arow = attn + h * 1024 + i * 32;
#pragma unroll
    for (int j = 0; j < LINE; ++j) arow[j] = (_Float16)(v[j] * inv);
#pragma unroll
    for (int j = LINE; j < 32; ++j) arow[j] = (_Float16)0.f;
  } else {
    const int u = tid - HEADS * LINE;           // 0..63 -> zero pad rows 24..31 of each head
    const int h = u >> 3, i = LINE + (u & 7);
    _Float16* arow = attn + h * 1024 + i * 32;
#pragma unroll
    for (int j = 0; j < 32; ++j) arow[j] = (_Float16)0.f;
  }
  __syncthreads();

  // ---- Phase 5: out = attn . v per head -> outb [token][h*64+d] f16 ----
  {
    const int h = wv;
    const _Float16* ah = attn + h * 1024;
#pragma unroll
    for (int mt = 0; mt < 2; ++mt)
#pragma unroll
      for (int nd = 0; nd < 4; ++nd) {
        v16h a = ldA(ah + (16 * mt + lm) * 32 + 16 * lh);
        v16h b = ldB(vT + (64 * h + 16 * nd + lm) * VT_PITCH + 16 * lh);
        v8f acc = vzero();
        acc = wmma(a, b, acc);
#pragma unroll
        for (int r = 0; r < 8; ++r)
          outb[(16 * mt + 8 * lh + r) * QO_PITCH + 64 * h + 16 * nd + lm] = (_Float16)acc[r];
      }
  }
  __syncthreads();

  // ---- Phase 6: y = out . Wo + bo  ([32x512] x [512x256]) ----
  // k-blocked: A block (2 Mtiles x 8 ksteps) register-resident, accumulators for
  // both N-tiles persist across the two k-blocks.
  {
    float* yline = y + (size_t)blockIdx.x * (LINE * DIM);
    v8f acc[2][2] = {{vzero(), vzero()}, {vzero(), vzero()}};   // [i2][m]
    const _Float16* wrow0 = Woh + ((wv)     * 16 + lm) * 512 + 16 * lh;
    const _Float16* wrow1 = Woh + ((wv + 8) * 16 + lm) * 512 + 16 * lh;
#pragma unroll
    for (int kblk = 0; kblk < 2; ++kblk) {
      v16h areg[2][8];
#pragma unroll
      for (int m = 0; m < 2; ++m)
#pragma unroll
        for (int ks = 0; ks < 8; ++ks)
          areg[m][ks] = ldA(outb + (16 * m + lm) * QO_PITCH + 256 * kblk + 32 * ks + 16 * lh);
#pragma unroll
      for (int i2 = 0; i2 < 2; ++i2) {
        const _Float16* wr = ((i2 == 0) ? wrow0 : wrow1) + 256 * kblk;
#pragma unroll
        for (int ks = 0; ks < 8; ++ks) {
          v16h b = ldB(wr + 32 * ks);
          acc[i2][0] = wmma(areg[0][ks], b, acc[i2][0]);
          acc[i2][1] = wmma(areg[1][ks], b, acc[i2][1]);
        }
      }
    }
#pragma unroll
    for (int i2 = 0; i2 < 2; ++i2) {
      const int nt = wv + 8 * i2;
      const float bias = bo[nt * 16 + lm];
#pragma unroll
      for (int r = 0; r < 8; ++r) {
        int row0 = 8 * lh + r;
        if (row0 < LINE) yline[row0 * DIM + nt * 16 + lm] = acc[i2][0][r] + bias;
        int row1 = 16 + 8 * lh + r;
        if (row1 < LINE) yline[row1 * DIM + nt * 16 + lm] = acc[i2][1][r] + bias;
      }
    }
  }
}

extern "C" void kernel_launch(void* const* d_in, const int* in_sizes, int n_in,
                              void* d_out, int out_size, void* d_ws, size_t ws_size,
                              hipStream_t stream) {
  (void)n_in; (void)out_size; (void)ws_size;
  const float* x   = (const float*)d_in[0];
  const float* Wq  = (const float*)d_in[1];
  const float* Wkv = (const float*)d_in[2];
  const float* Wo  = (const float*)d_in[3];
  const float* bo  = (const float*)d_in[4];
  const float* pos = (const float*)d_in[5];

  _Float16* W1h = (_Float16*)d_ws;              // [1536][256] f16
  _Float16* Woh = W1h + 1536 * 256;             // [256][512]  f16

  prep_kernel<<<2048, 256, 0, stream>>>(Wq, Wkv, Wo, W1h, Woh);

  const int nTok   = in_sizes[0] / DIM;
  const int nLines = nTok / LINE;
  (void)hipFuncSetAttribute((const void*)line_attn_kernel,
                            hipFuncAttributeMaxDynamicSharedMemorySize, SMEM_BYTES);
  line_attn_kernel<<<nLines, 256, SMEM_BYTES, stream>>>(x, bo, pos, W1h, Woh, (float*)d_out);
}
